// Attention_74028056314211
// MI455X (gfx1250) — compile-verified, matrix operand order checked by reading, not run on hardware
//
#include <hip/hip_runtime.h>
#include <hip/hip_bf16.h>

typedef _Float16 h16;
typedef __attribute__((ext_vector_type(16))) _Float16 v16h;
typedef __attribute__((ext_vector_type(8)))  float    v8f;

#define WMMA_F32_F16(a, b, c) \
  __builtin_amdgcn_wmma_f32_16x16x32_f16(false, (a), false, (b), (short)0, (c), false, false)

static constexpr int Bc = 2, Sq = 2048, Dm = 1024, Hh = 16, HD = 64;

// Build a 16x32 f16 fragment (A-matrix layout, or a column of B read as B^T rows).
// ISA 16-bit A layout: lane L -> row = L&15, half = L>>4;
//   VGPR 0..3: K = 2*i + half*8 + {0,1};  VGPR 4..7: K = 16 + 2*(i-4) + half*8 + {0,1}
// K-pairs are contiguous -> 8 aligned u32 loads per lane (compiler fuses to ds_load_b128).
__device__ __forceinline__ v16h load_frag(const h16* tile, int stride, int lane) {
  const int row = lane & 15, half = lane >> 4;
  union { v16h v; unsigned u[8]; } f;
  const unsigned* p = (const unsigned*)(tile + (size_t)row * stride);
#pragma unroll
  for (int i = 0; i < 8; ++i) {
    const int k = (i < 4 ? 2 * i : 16 + 2 * (i - 4)) + half * 8;
    f.u[i] = p[k >> 1];
  }
  return f.v;
}

// Async global->LDS copy of 16 bytes (ASYNCcnt-tracked, bypasses VGPRs).
// VDST VGPR = per-lane LDS byte offset (low 32 bits of the shared pointer per
// the flat-aperture truncation rule); VADDR = 64-bit global address.
__device__ __forceinline__ void async_copy_b128(h16* lds_dst, const h16* gsrc) {
  unsigned loff = (unsigned)(size_t)lds_dst;
  unsigned long long ga = (unsigned long long)(size_t)gsrc;
  __asm__ volatile("global_load_async_to_lds_b128 %0, %1, off"
                   :: "v"(loff), "v"(ga) : "memory");
}
__device__ __forceinline__ void wait_asynccnt0() {
  __asm__ volatile("s_wait_asynccnt 0x0" ::: "memory");
}

// ---------------------------------------------------------------------------
// Kernel 1: qkv = hidden @ W_attn + b_attn ; scatter to q/k [B,H,S,64] (q pre-
// scaled by 1/8) and vT [B,H,64,S], all f16.  128x128 tile, 8 waves (4Mx2N),
// each wave = 2x4 grid of 16x16 WMMA tiles, K stepped by 32 through LDS.
// ---------------------------------------------------------------------------
__global__ void __launch_bounds__(256)
qkv_gemm_kernel(const float* __restrict__ X, const float* __restrict__ W,
                const float* __restrict__ bias,
                h16* __restrict__ qws, h16* __restrict__ kws, h16* __restrict__ vT) {
  __shared__ h16 As[128 * 40];   // [m][k] pad 40
  __shared__ h16 Bs[128 * 40];   // [n][k] (transposed) pad 40
  const int tid = threadIdx.x;
  const int lane = tid & 31, wave = tid >> 5;
  const int wm = wave & 3, wn = wave >> 2;
  const int m0 = blockIdx.y * 128, n0 = blockIdx.x * 128;

  v8f acc[2][4];
#pragma unroll
  for (int i = 0; i < 2; ++i)
#pragma unroll
    for (int j = 0; j < 4; ++j) acc[i][j] = (v8f)0.f;

  const int arow = tid >> 1;          // 0..127
  const int ac4  = (tid & 1) * 4;     // float4 base in 32-wide K slab
  const int brow = tid >> 3;          // 0..31  (k within slab)
  const int bc4  = tid & 7;

  for (int kt = 0; kt < Dm; kt += 32) {
    __syncthreads();
    { // stage A (f32 -> f16)
      const float4* src = (const float4*)(X + (size_t)(m0 + arow) * Dm + kt);
#pragma unroll
      for (int j = 0; j < 4; ++j) {
        float4 v = src[ac4 + j];
        h16* dst = As + arow * 40 + (ac4 + j) * 4;
        dst[0] = (h16)v.x; dst[1] = (h16)v.y; dst[2] = (h16)v.z; dst[3] = (h16)v.w;
      }
    }
    { // stage B transposed (f32 -> f16)
      const float4* src = (const float4*)(W + (size_t)(kt + brow) * (3 * Dm) + n0);
#pragma unroll
      for (int j = 0; j < 4; ++j) {
        const int c4 = bc4 + 8 * j;
        float4 v = src[c4];
        Bs[(c4 * 4 + 0) * 40 + brow] = (h16)v.x;
        Bs[(c4 * 4 + 1) * 40 + brow] = (h16)v.y;
        Bs[(c4 * 4 + 2) * 40 + brow] = (h16)v.z;
        Bs[(c4 * 4 + 3) * 40 + brow] = (h16)v.w;
      }
    }
    if (kt + 32 < Dm) {  // global_prefetch_b8 next slabs
      __builtin_prefetch(X + (size_t)(m0 + arow) * Dm + kt + 32, 0, 1);
      __builtin_prefetch(W + (size_t)(kt + 32 + brow) * (3 * Dm) + n0, 0, 1);
    }
    __syncthreads();
    v16h af[2], bf[4];
#pragma unroll
    for (int mi = 0; mi < 2; ++mi) af[mi] = load_frag(As + (wm * 32 + mi * 16) * 40, 40, lane);
#pragma unroll
    for (int ni = 0; ni < 4; ++ni) bf[ni] = load_frag(Bs + (wn * 64 + ni * 16) * 40, 40, lane);
#pragma unroll
    for (int mi = 0; mi < 2; ++mi)
#pragma unroll
      for (int ni = 0; ni < 4; ++ni)
        acc[mi][ni] = WMMA_F32_F16(af[mi], bf[ni], acc[mi][ni]);
  }

  // epilogue: bias + scatter (block-uniform q/k/v branch since 128 | 1024)
  const int half = lane >> 4, col = lane & 15;
#pragma unroll
  for (int mi = 0; mi < 2; ++mi)
#pragma unroll
    for (int ni = 0; ni < 4; ++ni) {
      const int n = n0 + wn * 64 + ni * 16 + col;
      const float bv = bias[n];
      const int which = n >> 10, nn = n & 1023;
      const int h = nn >> 6, d = nn & 63;
#pragma unroll
      for (int i = 0; i < 8; ++i) {
        const int m = m0 + wm * 32 + mi * 16 + i + half * 8;
        const int b = m >> 11, s = m & 2047;
        const int bh = b * Hh + h;
        const float v = acc[mi][ni][i] + bv;
        if (which == 0)
          qws[((size_t)bh * Sq + s) * HD + d] = (h16)(v * 0.125f);  // 1/sqrt(64)
        else if (which == 1)
          kws[((size_t)bh * Sq + s) * HD + d] = (h16)v;
        else
          vT[((size_t)bh * HD + d) * Sq + s] = (h16)v;              // transposed
      }
    }
}

// ---------------------------------------------------------------------------
// Kernel 2: causal flash attention. One wave per 16-query tile per (b,h).
// Score chunk = 32 keys: 4 WMMAs (K=64 split 2x32) -> online softmax ->
// P tile through per-wave LDS (s_wait_dscnt) -> 4 PV WMMAs (K=32).
// ---------------------------------------------------------------------------
__global__ void __launch_bounds__(256)
flash_attn_kernel(const h16* __restrict__ qws, const h16* __restrict__ kws,
                  const h16* __restrict__ vT, h16* __restrict__ merged) {
  __shared__ h16 P[8][16 * 40];
  const int lane = threadIdx.x & 31, warp = threadIdx.x >> 5;
  const int wave = blockIdx.x * 8 + warp;
  const int qt = wave & 127;          // S/16 = 128 q-tiles
  const int bh = wave >> 7;           // 0..31
  const int q0 = qt * 16;
  const int half = lane >> 4, col = lane & 15;

  const h16* qbase = qws + ((size_t)bh * Sq + q0) * HD;
  v16h aq[2];
#pragma unroll
  for (int kk = 0; kk < 2; ++kk) aq[kk] = load_frag(qbase + kk * 32, HD, lane);

  float rm[8], rs[8];
#pragma unroll
  for (int i = 0; i < 8; ++i) { rm[i] = -1e30f; rs[i] = 0.f; }
  v8f accO[4];
#pragma unroll
  for (int n = 0; n < 4; ++n) accO[n] = (v8f)0.f;

  h16* Pw = &P[warp][0];

  for (int k0 = 0; k0 < q0 + 16; k0 += 32) {     // never past the diagonal
    const h16* kbase = kws + ((size_t)bh * Sq + k0) * HD;
    v8f sc[2];
#pragma unroll
    for (int t = 0; t < 2; ++t) {
      v8f s = (v8f)0.f;
#pragma unroll
      for (int kk = 0; kk < 2; ++kk) {
        v16h bk = load_frag(kbase + t * 16 * HD + kk * 32, HD, lane);
        s = WMMA_F32_F16(aq[kk], bk, s);
      }
      sc[t] = s;
    }
    if (k0 + 31 > q0) {  // diagonal chunk: causal mask
#pragma unroll
      for (int t = 0; t < 2; ++t) {
        const int kg = k0 + t * 16 + col;
#pragma unroll
        for (int i = 0; i < 8; ++i)
          if (kg > q0 + i + half * 8) sc[t][i] = -1e30f;
      }
    }
    // online softmax: rows live as vgpr index i (+ half*8); cols across 16 lanes
#pragma unroll
    for (int i = 0; i < 8; ++i) {
      float m = fmaxf(sc[0][i], sc[1][i]);
      m = fmaxf(m, __shfl_xor(m, 1));
      m = fmaxf(m, __shfl_xor(m, 2));
      m = fmaxf(m, __shfl_xor(m, 4));
      m = fmaxf(m, __shfl_xor(m, 8));
      const float mnew = fmaxf(rm[i], m);
      const float corr = __expf(rm[i] - mnew);
      rm[i] = mnew;
      const float p0 = __expf(sc[0][i] - mnew);
      const float p1 = __expf(sc[1][i] - mnew);
      sc[0][i] = p0; sc[1][i] = p1;
      float ls = p0 + p1;
      ls += __shfl_xor(ls, 1);
      ls += __shfl_xor(ls, 2);
      ls += __shfl_xor(ls, 4);
      ls += __shfl_xor(ls, 8);
      rs[i] = rs[i] * corr + ls;
#pragma unroll
      for (int n = 0; n < 4; ++n) accO[n][i] *= corr;
    }
    // C-layout -> A-layout for P via per-wave LDS tile
#pragma unroll
    for (int t = 0; t < 2; ++t)
#pragma unroll
      for (int i = 0; i < 8; ++i)
        Pw[(i + half * 8) * 40 + t * 16 + col] = (h16)sc[t][i];
    __asm__ volatile("s_wait_dscnt 0" ::: "memory");
    v16h ap = load_frag(Pw, 40, lane);
#pragma unroll
    for (int n = 0; n < 4; ++n) {
      const h16* vbase = vT + ((size_t)bh * HD + n * 16) * Sq + k0;  // B^T rows = d-cols
      v16h bv = load_frag(vbase, Sq, lane);
      accO[n] = WMMA_F32_F16(ap, bv, accO[n]);
    }
  }

  const int b = bh >> 4, h = bh & 15;
#pragma unroll
  for (int i = 0; i < 8; ++i) {
    const float inv = 1.f / rs[i];
    const int s = q0 + i + half * 8;
#pragma unroll
    for (int n = 0; n < 4; ++n) {
      const int d = h * HD + n * 16 + col;
      merged[((size_t)b * Sq + s) * Dm + d] = (h16)(accO[n][i] * inv);
    }
  }
}

// ---------------------------------------------------------------------------
// Kernel 3: out = merged(f16) @ W_proj + b_proj, f32 output.
// A-tile staging is a pure f16 copy -> use async global->LDS (ASYNCcnt path).
// ---------------------------------------------------------------------------
__global__ void __launch_bounds__(256)
proj_gemm_kernel(const h16* __restrict__ Xh, const float* __restrict__ W,
                 const float* __restrict__ bias, float* __restrict__ out) {
  __shared__ h16 As[128 * 40];
  __shared__ h16 Bs[128 * 40];
  const int tid = threadIdx.x;
  const int lane = tid & 31, wave = tid >> 5;
  const int wm = wave & 3, wn = wave >> 2;
  const int m0 = blockIdx.y * 128, n0 = blockIdx.x * 128;

  v8f acc[2][4];
#pragma unroll
  for (int i = 0; i < 2; ++i)
#pragma unroll
    for (int j = 0; j < 4; ++j) acc[i][j] = (v8f)0.f;

  const int arow = tid >> 1;
  const int ah   = (tid & 1) * 16;  // h16 element base within the 32-wide K slab
  const int brow = tid >> 3;
  const int bc4  = tid & 7;

  for (int kt = 0; kt < Dm; kt += 32) {
    __syncthreads();
    { // A already f16: async copy straight into LDS (2 x b128 per thread)
      const h16* gsrc = Xh + (size_t)(m0 + arow) * Dm + kt + ah;
      h16* ldst = As + arow * 40 + ah;
      async_copy_b128(ldst, gsrc);
      async_copy_b128(ldst + 8, gsrc + 8);
    }
    { // B f32 -> f16, transposed
      const float4* src = (const float4*)(W + (size_t)(kt + brow) * Dm + n0);
#pragma unroll
      for (int j = 0; j < 4; ++j) {
        const int c4 = bc4 + 8 * j;
        float4 v = src[c4];
        Bs[(c4 * 4 + 0) * 40 + brow] = (h16)v.x;
        Bs[(c4 * 4 + 1) * 40 + brow] = (h16)v.y;
        Bs[(c4 * 4 + 2) * 40 + brow] = (h16)v.z;
        Bs[(c4 * 4 + 3) * 40 + brow] = (h16)v.w;
      }
    }
    if (kt + 32 < Dm) {
      __builtin_prefetch(Xh + (size_t)(m0 + arow) * Dm + kt + 32, 0, 1);
      __builtin_prefetch(W + (size_t)(kt + 32 + brow) * Dm + n0, 0, 1);
    }
    wait_asynccnt0();   // async LDS writes visible before the barrier
    __syncthreads();
    v16h af[2], bf[4];
#pragma unroll
    for (int mi = 0; mi < 2; ++mi) af[mi] = load_frag(As + (wm * 32 + mi * 16) * 40, 40, lane);
#pragma unroll
    for (int ni = 0; ni < 4; ++ni) bf[ni] = load_frag(Bs + (wn * 64 + ni * 16) * 40, 40, lane);
#pragma unroll
    for (int mi = 0; mi < 2; ++mi)
#pragma unroll
      for (int ni = 0; ni < 4; ++ni)
        acc[mi][ni] = WMMA_F32_F16(af[mi], bf[ni], acc[mi][ni]);
  }

  const int half = lane >> 4, col = lane & 15;
#pragma unroll
  for (int mi = 0; mi < 2; ++mi)
#pragma unroll
    for (int ni = 0; ni < 4; ++ni) {
      const int n = n0 + wn * 64 + ni * 16 + col;
      const float bv = bias[n];
#pragma unroll
      for (int i = 0; i < 8; ++i) {
        const int m = m0 + wm * 32 + mi * 16 + i + half * 8;
        out[(size_t)m * Dm + n] = acc[mi][ni][i] + bv;
      }
    }
}

extern "C" void kernel_launch(void* const* d_in, const int* in_sizes, int n_in,
                              void* d_out, int out_size, void* d_ws, size_t ws_size,
                              hipStream_t stream) {
  (void)in_sizes; (void)n_in; (void)out_size; (void)ws_size;
  const float* hidden = (const float*)d_in[0];
  const float* W_attn = (const float*)d_in[1];
  const float* b_attn = (const float*)d_in[2];
  const float* W_proj = (const float*)d_in[3];
  const float* b_proj = (const float*)d_in[4];
  float* out = (float*)d_out;

  h16* ws = (h16*)d_ws;
  const size_t per = (size_t)Bc * Hh * Sq * HD;  // 4,194,304 f16 elems each
  h16* qws    = ws;
  h16* kws    = ws + per;
  h16* vT     = ws + 2 * per;
  h16* merged = ws + 3 * per;                    // 32 MB total workspace

  dim3 blk(256);
  // QKV: M=4096 (32 tiles), N=3072 (24 tiles)
  qkv_gemm_kernel<<<dim3(24, 32), blk, 0, stream>>>(hidden, W_attn, b_attn, qws, kws, vT);
  // Attention: 2*16*(2048/16) = 4096 waves / 8 per block = 512 blocks
  flash_attn_kernel<<<dim3(512), blk, 0, stream>>>(qws, kws, vT, merged);
  // Proj: M=4096 (32 tiles), N=1024 (8 tiles)
  proj_gemm_kernel<<<dim3(8, 32), blk, 0, stream>>>(merged, W_proj, b_proj, out);
}
